// CausalSelfAttention_26328149524946
// MI455X (gfx1250) — matrix operand order repaired and checked by reading, not compiled, not run on hardware
//
#include <hip/hip_runtime.h>

#define Bsz 4
#define Tsz 2048
#define Csz 1024
#define Hn  16
#define Dh  64

typedef __attribute__((ext_vector_type(16))) _Float16 v16h;
typedef __attribute__((ext_vector_type(8)))  _Float16 v8h;   // 16 bytes
typedef __attribute__((ext_vector_type(4)))  _Float16 v4h;   // 8 bytes
typedef __attribute__((ext_vector_type(8)))  float    v8f;

__device__ __forceinline__ v8f wmma_f16(v16h a, v16h b, v8f c) {
  return __builtin_amdgcn_wmma_f32_16x16x32_f16(false, a, false, b, (short)0, c, false, false);
}

__device__ __forceinline__ v16h ld_frag16(const _Float16* p0, const _Float16* p1) {
  v8h lo = *(const v8h*)p0;
  v8h hi = *(const v8h*)p1;
  return __builtin_shufflevector(lo, hi, 0,1,2,3,4,5,6,7,8,9,10,11,12,13,14,15);
}

// A-fragment 16x32 f16 from row-major p[m*stride + k]; 2 x b128.
__device__ __forceinline__ v16h frag_a(const _Float16* p, int stride, int lane) {
  const _Float16* row = p + (lane & 15) * stride + (lane >> 4) * 8;
  return ld_frag16(row, row + 16);
}

// B-fragment 32x16 f16 from N-major (transposed) tile p[n*stride + k]; 2 x b128.
__device__ __forceinline__ v16h frag_bt(const _Float16* p, int stride, int lane) {
  const _Float16* row = p + (lane & 15) * stride + (lane >> 4) * 8;
  return ld_frag16(row, row + 16);
}

// ---------------- Kernel 1: fused QKV projection --------------------------
// x[8192,1024] f32 @ w[1024,3072] f32 + bias -> Q/K/V f16 in [B,H,T,D]
// 128 threads = 4 waves; block tile 128m x 64n; wave tile 32m x 64n.
__global__ __launch_bounds__(128) void k_gemm_qkv(
    const float* __restrict__ x, const float* __restrict__ w,
    const float* __restrict__ bias,
    _Float16* __restrict__ qb, _Float16* __restrict__ kb, _Float16* __restrict__ vb)
{
  __shared__ __align__(16) _Float16 As[128][40];  // [m][k]
  __shared__ __align__(16) _Float16 Wt[64][40];   // [n][k]  (transposed tile)
  const int tid = threadIdx.x, lane = tid & 31, wave = tid >> 5;
  const int mbase = blockIdx.y * 128, nbase = blockIdx.x * 64;
  v8f acc[2][4] = {};

  for (int k0 = 0; k0 < Csz; k0 += 32) {
#pragma unroll
    for (int i = 0; i < 8; ++i) {            // A tile: 128x32 f32 -> f16
      int g = tid + i * 128;                 // 1024 groups
      int m = g >> 3, kk = (g & 7) * 4;
      float4 f = *(const float4*)(x + (size_t)(mbase + m) * Csz + k0 + kk);
      v4h hv = { (_Float16)f.x, (_Float16)f.y, (_Float16)f.z, (_Float16)f.w };
      *(v4h*)&As[m][kk] = hv;
    }
#pragma unroll
    for (int i = 0; i < 4; ++i) {            // W tile transposed (coalesced in n)
      int g = tid + i * 128;                 // 512 groups
      int n = g & 63, kk = (g >> 6) * 4;
      const float* col = w + (size_t)(k0 + kk) * (3 * Csz) + nbase + n;
      v4h hv = { (_Float16)col[0], (_Float16)col[3 * Csz],
                 (_Float16)col[6 * Csz], (_Float16)col[9 * Csz] };
      *(v4h*)&Wt[n][kk] = hv;
    }
    __syncthreads();
    v16h a0 = frag_a(&As[wave * 32][0],      40, lane);
    v16h a1 = frag_a(&As[wave * 32 + 16][0], 40, lane);
#pragma unroll
    for (int j = 0; j < 4; ++j) {
      v16h bf = frag_bt(&Wt[j * 16][0], 40, lane);
      acc[0][j] = wmma_f16(a0, bf, acc[0][j]);
      acc[1][j] = wmma_f16(a1, bf, acc[1][j]);
    }
    __syncthreads();
  }

  // Epilogue: nbase is 64-aligned -> q/k/v select and head are block constants.
  const int hi = lane >> 4, n = lane & 15;
  const int which = nbase >> 10;
  const int h     = (nbase & 1023) >> 6;
  _Float16* __restrict__ dstbuf = (which == 0) ? qb : (which == 1) ? kb : vb;
#pragma unroll
  for (int mi = 0; mi < 2; ++mi) {
#pragma unroll
    for (int j = 0; j < 4; ++j) {
      float bv = bias[nbase + j * 16 + n];
#pragma unroll
      for (int r = 0; r < 8; ++r) {
        int gm = mbase + wave * 32 + mi * 16 + r + hi * 8;
        int b = gm >> 11, t = gm & (Tsz - 1);
        size_t dst = (((size_t)(b * Hn + h)) * Tsz + t) * Dh + j * 16 + n;
        dstbuf[dst] = (_Float16)(acc[mi][j][r] + bv);
      }
    }
  }
}

// ---------------- Kernel 2: causal flash attention ------------------------
// grid (T/64, B*H), 128 threads (4 waves, 16 q-rows each)
__global__ __launch_bounds__(128) void k_attn(
    const _Float16* __restrict__ qg, const _Float16* __restrict__ kg,
    const _Float16* __restrict__ vg, _Float16* __restrict__ y)
{
  __shared__ __align__(16) _Float16 Ks[32][72];      // [key][d]
  __shared__ __align__(16) _Float16 Vt[64][40];      // [d][key] (transposed)
  __shared__ __align__(16) _Float16 Ps[4][16][40];   // per-wave P scratch

  const int tid = threadIdx.x, lane = tid & 31, wave = tid >> 5;
  const int bh = blockIdx.y, b = bh >> 4, h = bh & 15;
  const int qblk  = blockIdx.x * 64;
  const int qbase = qblk + wave * 16;
  const _Float16* Qp = qg + (size_t)bh * Tsz * Dh;
  const _Float16* Kp = kg + (size_t)bh * Tsz * Dh;
  const _Float16* Vp = vg + (size_t)bh * Tsz * Dh;
  const int n = lane & 15, hi = lane >> 4;

  v16h qa0 = frag_a(Qp + (size_t)qbase * Dh, Dh, lane);        // d 0..31
  v16h qa1 = frag_a(Qp + (size_t)qbase * Dh + 32, Dh, lane);   // d 32..63

  float mrun[8], lrun[8];
#pragma unroll
  for (int r = 0; r < 8; ++r) { mrun[r] = -__builtin_inff(); lrun[r] = 0.f; }
  v8f o0 = {}, o1 = {}, o2 = {}, o3 = {};

  const int kend = qblk + 64;
  for (int kt = 0; kt < kend; kt += 32) {
#pragma unroll
    for (int i = 0; i < 4; ++i) {            // K tile: v4h copies
      int g = tid + i * 128;                 // 512 groups
      int r = g >> 4, c = (g & 15) * 4;
      *(v4h*)&Ks[r][c] = *(const v4h*)(Kp + (size_t)(kt + r) * Dh + c);
    }
#pragma unroll
    for (int i = 0; i < 4; ++i) {            // V tile transposed (coalesced in d)
      int g = tid + i * 128;                 // 512 groups
      int d = g & 63, kk = (g >> 6) * 4;
      const _Float16* col = Vp + (size_t)(kt + kk) * Dh + d;
      v4h hv = { col[0], col[Dh], col[2 * Dh], col[3 * Dh] };
      *(v4h*)&Vt[d][kk] = hv;
    }
    __syncthreads();

    // S = Q Kᵀ : B-frag cols = keys, matmul-K = d -> Ks rows are K-contiguous
    v8f s0 = {}, s1 = {};
    s0 = wmma_f16(qa0, frag_bt(&Ks[0][0],   72, lane), s0);
    s0 = wmma_f16(qa1, frag_bt(&Ks[0][32],  72, lane), s0);
    s1 = wmma_f16(qa0, frag_bt(&Ks[16][0],  72, lane), s1);
    s1 = wmma_f16(qa1, frag_bt(&Ks[16][32], 72, lane), s1);

    // online softmax (each output row lives on one 16-lane half)
#pragma unroll
    for (int r = 0; r < 8; ++r) {
      int grow = qbase + r + hi * 8;
      float a0 = s0[r] * 0.125f;
      float a1 = s1[r] * 0.125f;
      if (kt + n > grow)      a0 = -__builtin_inff();
      if (kt + 16 + n > grow) a1 = -__builtin_inff();
      float rm = fmaxf(a0, a1);
#pragma unroll
      for (int off = 1; off < 16; off <<= 1) rm = fmaxf(rm, __shfl_xor(rm, off, 32));
      float mn  = fmaxf(mrun[r], rm);
      float fac = __expf(mrun[r] - mn);
      mrun[r] = mn;
      float p0 = __expf(a0 - mn), p1 = __expf(a1 - mn);
      float rs = p0 + p1;
#pragma unroll
      for (int off = 1; off < 16; off <<= 1) rs += __shfl_xor(rs, off, 32);
      lrun[r] = lrun[r] * fac + rs;
      o0[r] = o0[r] * fac; o1[r] = o1[r] * fac;
      o2[r] = o2[r] * fac; o3[r] = o3[r] * fac;
      Ps[wave][r + hi * 8][n]      = (_Float16)p0;
      Ps[wave][r + hi * 8][16 + n] = (_Float16)p1;
    }
    asm volatile("" ::: "memory");  // per-wave LDS is in-order; compiler fence only

    // O += P V
    v16h pa = frag_a(&Ps[wave][0][0], 40, lane);
    o0 = wmma_f16(pa, frag_bt(&Vt[0][0],  40, lane), o0);
    o1 = wmma_f16(pa, frag_bt(&Vt[16][0], 40, lane), o1);
    o2 = wmma_f16(pa, frag_bt(&Vt[32][0], 40, lane), o2);
    o3 = wmma_f16(pa, frag_bt(&Vt[48][0], 40, lane), o3);
    __syncthreads();
  }

  // normalize + store y[B,T,C] f16 (c = h*64 + d)
#pragma unroll
  for (int r = 0; r < 8; ++r) {
    int grow = qbase + r + hi * 8;
    float inv = 1.0f / lrun[r];
    size_t base = ((size_t)(b * Tsz + grow)) * Csz + h * Dh;
    y[base +      n] = (_Float16)(o0[r] * inv);
    y[base + 16 + n] = (_Float16)(o1[r] * inv);
    y[base + 32 + n] = (_Float16)(o2[r] * inv);
    y[base + 48 + n] = (_Float16)(o3[r] * inv);
  }
}

// ---------------- Kernel 3: output projection -----------------------------
// y[8192,1024] f16 @ w_proj[1024,1024] f32 + bias -> out f32
__global__ __launch_bounds__(128) void k_gemm_proj(
    const _Float16* __restrict__ y, const float* __restrict__ w,
    const float* __restrict__ bias, float* __restrict__ out)
{
  __shared__ __align__(16) _Float16 As[128][40];  // [m][k]
  __shared__ __align__(16) _Float16 Wt[64][40];   // [n][k]
  const int tid = threadIdx.x, lane = tid & 31, wave = tid >> 5;
  const int mbase = blockIdx.y * 128, nbase = blockIdx.x * 64;
  v8f acc[2][4] = {};

  for (int k0 = 0; k0 < Csz; k0 += 32) {
#pragma unroll
    for (int i = 0; i < 8; ++i) {            // A tile: 128x32 f16 copies
      int g = tid + i * 128;                 // 1024 groups
      int m = g >> 3, kk = (g & 7) * 4;
      *(v4h*)&As[m][kk] = *(const v4h*)(y + (size_t)(mbase + m) * Csz + k0 + kk);
    }
#pragma unroll
    for (int i = 0; i < 4; ++i) {            // W tile transposed
      int g = tid + i * 128;                 // 512 groups
      int n = g & 63, kk = (g >> 6) * 4;
      const float* col = w + (size_t)(k0 + kk) * Csz + nbase + n;
      v4h hv = { (_Float16)col[0], (_Float16)col[Csz],
                 (_Float16)col[2 * Csz], (_Float16)col[3 * Csz] };
      *(v4h*)&Wt[n][kk] = hv;
    }
    __syncthreads();
    v16h a0 = frag_a(&As[wave * 32][0],      40, lane);
    v16h a1 = frag_a(&As[wave * 32 + 16][0], 40, lane);
#pragma unroll
    for (int j = 0; j < 4; ++j) {
      v16h bf = frag_bt(&Wt[j * 16][0], 40, lane);
      acc[0][j] = wmma_f16(a0, bf, acc[0][j]);
      acc[1][j] = wmma_f16(a1, bf, acc[1][j]);
    }
    __syncthreads();
  }

  const int hi = lane >> 4, n = lane & 15;
#pragma unroll
  for (int mi = 0; mi < 2; ++mi) {
#pragma unroll
    for (int j = 0; j < 4; ++j) {
      float bv = bias[nbase + j * 16 + n];
#pragma unroll
      for (int r = 0; r < 8; ++r) {
        int gm = mbase + wave * 32 + mi * 16 + r + hi * 8;
        out[(size_t)gm * Csz + nbase + j * 16 + n] = acc[mi][j][r] + bv;
      }
    }
  }
}

// ---------------- Host launcher -------------------------------------------
extern "C" void kernel_launch(void* const* d_in, const int* in_sizes, int n_in,
                              void* d_out, int out_size, void* d_ws, size_t ws_size,
                              hipStream_t stream) {
  const float* x      = (const float*)d_in[0];
  const float* w_attn = (const float*)d_in[1];
  const float* b_attn = (const float*)d_in[2];
  const float* w_proj = (const float*)d_in[3];
  const float* b_proj = (const float*)d_in[4];
  float* out = (float*)d_out;

  const size_t NE = (size_t)Bsz * Hn * Tsz * Dh;   // 8,388,608 elements
  _Float16* qb = (_Float16*)d_ws;
  _Float16* kb = qb + NE;
  _Float16* vb = kb + NE;
  _Float16* yb = vb + NE;                           // total 64 MB f16

  dim3 g1(3 * Csz / 64, (Bsz * Tsz) / 128);         // (48, 64)
  k_gemm_qkv<<<g1, 128, 0, stream>>>(x, w_attn, b_attn, qb, kb, vb);

  dim3 g2(Tsz / 64, Bsz * Hn);                      // (32, 64)
  k_attn<<<g2, 128, 0, stream>>>(qb, kb, vb, yb);

  dim3 g3(Csz / 64, (Bsz * Tsz) / 128);             // (16, 64)
  k_gemm_proj<<<g3, 128, 0, stream>>>(yb, w_proj, b_proj, out);
}
